// StateSpaceModel_Cell_62019327754327
// MI455X (gfx1250) — matrix operand order repaired and checked by reading, not compiled, not run on hardware
//
#include <hip/hip_runtime.h>

// ---------------------------------------------------------------------------
// Problem constants (from reference): B=16384, T=10, F=34, E=H=256, fut=9
// ---------------------------------------------------------------------------
#define BSZ   16384
#define TT    10
#define FF    34
#define HH    256
#define G4    1024  // 4*H gates
#define FUT   9

typedef __attribute__((ext_vector_type(8)))  __bf16 v8bf;
typedef __attribute__((ext_vector_type(16))) __bf16 v16bf;
typedef __attribute__((ext_vector_type(8)))  float  v8f;
typedef __attribute__((ext_vector_type(4)))  unsigned int u32x4;
typedef __attribute__((ext_vector_type(8)))  int i32x8;
typedef __attribute__((ext_vector_type(4)))  int i32x4;

__device__ __forceinline__ float sigf(float x) { return 1.0f / (1.0f + __expf(-x)); }

// ---------------------------------------------------------------------------
// TDM: DMA one weight slice (32 rows x 512 B, row stride 512 B) into LDS.
// D# per CDNA5 ISA. data_size = 8 bytes (code 3), so all
// dims/strides are in 8-byte units: row = 64 units, 32 rows, stride 64.
// Toolchain uses the 6-arg builtin: (g0, g1, g2, g3, ext, cpol).
// ---------------------------------------------------------------------------
__device__ __forceinline__ void tdm_load_rows(const __bf16* gsrc, unsigned int lds_byte_addr)
{
    unsigned long long ga = (unsigned long long)(uintptr_t)gsrc;
    u32x4 g0;
    g0[0] = 1u;                                  // count=1 valid descriptor
    g0[1] = lds_byte_addr;                       // lds_addr [63:32]
    g0[2] = (unsigned int)(ga & 0xFFFFFFFFu);    // global_addr[31:0]
    g0[3] = (unsigned int)((ga >> 32) & 0x01FFFFFFu) | (2u << 30); // addr[56:32] | type=2
    i32x8 g1;
    g1[0] = (3 << 16);        // workgroup_mask=0 | data_size=3 (8B) | no flags
    g1[1] = (64 << 16);       // [47:32] atomic_bar=0 | [63:48] tensor_dim0 lo16 = 64
    g1[2] = (32 << 16);       // [79:64] tensor_dim0 hi16 = 0 | [95:80] tensor_dim1 lo16 = 32
    g1[3] = (64 << 16);       // [111:96] tensor_dim1 hi16 = 0 | [127:112] tile_dim0 = 64
    g1[4] = 32;               // [143:128] tile_dim1 = 32 | [159:144] tile_dim2 = 0
    g1[5] = 64;               // [191:160] tensor_dim0_stride lo32 = 64
    g1[6] = 0;                // stride0 hi16 | tensor_dim1_stride lo16
    g1[7] = 0;                // tensor_dim1_stride hi32
    i32x4 z4 = {0, 0, 0, 0};             // groups 2/3 unused (2-D tensor)
    i32x8 z8 = {0, 0, 0, 0, 0, 0, 0, 0}; // extended operand unused
    __builtin_amdgcn_tensor_load_to_lds(g0, g1, z4, z4, z8, 0);
}

// ---------------------------------------------------------------------------
// Pack 4 LSTM weight matrices [1024x256] f32 -> bf16 (row-major kept: row n is
// gate-output column n, K contiguous == exact B-fragment feed order).
// ---------------------------------------------------------------------------
__global__ __launch_bounds__(256) void pack_weights(
    const float* __restrict__ w1, const float* __restrict__ w2,
    const float* __restrict__ w3, const float* __restrict__ w4,
    __bf16* __restrict__ o1, __bf16* __restrict__ o2,
    __bf16* __restrict__ o3, __bf16* __restrict__ o4)
{
    int i = blockIdx.x * 256 + threadIdx.x;
    if (i < G4 * HH) {
        o1[i] = (__bf16)w1[i];
        o2[i] = (__bf16)w2[i];
        o3[i] = (__bf16)w3[i];
        o4[i] = (__bf16)w4[i];
    }
}

__global__ __launch_bounds__(256) void combine_bias(
    const float* __restrict__ bih1, const float* __restrict__ bhh1,
    const float* __restrict__ bih2, const float* __restrict__ bhh2,
    float* __restrict__ b1, float* __restrict__ b2)
{
    int i = blockIdx.x * 256 + threadIdx.x;
    if (i < G4) {
        b1[i] = bih1[i] + bhh1[i];
        b2[i] = bih2[i] + bhh2[i];
    }
}

__global__ __launch_bounds__(256) void zero_state(
    __bf16* __restrict__ h1, __bf16* __restrict__ h2,
    float* __restrict__ c1, float* __restrict__ c2)
{
    int i = blockIdx.x * 256 + threadIdx.x; // covers BSZ*HH
    h1[i] = (__bf16)0.0f;
    h2[i] = (__bf16)0.0f;
    c1[i] = 0.0f;
    c2[i] = 0.0f;
}

// ---------------------------------------------------------------------------
// Encoder: enc[t][b][e] = bf16( x[b][t][:34] . Wenc[e][:34] + benc[e] )
// ---------------------------------------------------------------------------
__global__ __launch_bounds__(256) void encoder_kernel(
    const float* __restrict__ x, const float* __restrict__ Wenc,
    const float* __restrict__ benc, __bf16* __restrict__ enc)
{
    __shared__ float xs[FF];
    int row = blockIdx.x;          // 0 .. T*B-1
    int t = row >> 14;             // / BSZ
    int b = row & (BSZ - 1);
    int e = threadIdx.x;
    if (e < FF) xs[e] = x[((size_t)b * TT + t) * FF + e];
    __syncthreads();
    const float* wr = Wenc + (size_t)e * FF;
    float s = benc[e];
#pragma unroll
    for (int f = 0; f < FF; ++f) s += xs[f] * wr[f];
    enc[((size_t)t * BSZ + b) * HH + e] = (__bf16)s;
}

// ---------------------------------------------------------------------------
// LSTM cell: gates = X@Wihᵀ + Hin@Whhᵀ + bias; c/h update fused in registers.
//
// Grid (128, 8), 256 threads = 8 waves. Block tile: 128 batch rows x 32
// per-gate cols. Wave = (row-group 0..3, col-group 0..1); each wave owns TWO
// 16-row tiles so one set of B fragments feeds 8 WMMAs.
//
// Weights are DMA'd by the Tensor Data Mover: wave 0 issues 8
// tensor_load_to_lds at kernel entry (Wih slice -> LDS buf0, Whh slice ->
// LDS buf1, 64 KB each). s_wait_tensorcnt 4 gates phase 1 (TDM completes in
// order), so the Whh DMA overlaps phase-1 compute. 128 WMMA per wave.
// ---------------------------------------------------------------------------
__global__ __launch_bounds__(256) void lstm_cell(
    const __bf16* __restrict__ X,   const __bf16* __restrict__ Hin,
    __bf16* __restrict__ Hout,      float* __restrict__ C,
    const __bf16* __restrict__ Wih, const __bf16* __restrict__ Whh,
    const float* __restrict__ bias)
{
    extern __shared__ char smem[];
    __bf16* ldsW0 = (__bf16*)smem;            // Wih slice: [128][256] bf16, 64 KB
    __bf16* ldsW1 = ldsW0 + 128 * HH;         // Whh slice: 64 KB

    const int tid  = threadIdx.x;
    const int wave = tid >> 5;
    const int lane = tid & 31;
    const int lh   = lane >> 4;               // half-wave (K split)
    const int lm   = lane & 15;               // row (A) / col (B,C) within tile
    const int rg   = wave >> 1;               // 0..3
    const int cg   = wave & 1;                // 0..1
    const int m0a  = blockIdx.x * 128 + rg * 16;
    const int m0b  = m0a + 64;
    const int nblk = blockIdx.y * 32;         // per-gate col base of this block
    const int nl   = cg * 16 + lm;            // col within block slice (0..31)
    const int n    = nblk + nl;               // per-gate col (0..255)

    // ---- issue all weight DMA up front (wave 0 only, scalar branch) ----
    const int wave_u = __builtin_amdgcn_readfirstlane(wave);
    if (wave_u == 0) {
        const unsigned int lds0 = (unsigned int)(uintptr_t)ldsW0;
        const unsigned int lds1 = (unsigned int)(uintptr_t)ldsW1;
#pragma unroll
        for (int g = 0; g < 4; ++g)
            tdm_load_rows(Wih + (size_t)(g * HH + nblk) * HH, lds0 + g * 16384u);
#pragma unroll
        for (int g = 0; g < 4; ++g)
            tdm_load_rows(Whh + (size_t)(g * HH + nblk) * HH, lds1 + g * 16384u);
        __builtin_amdgcn_s_wait_tensorcnt(4);   // Wih group landed (in-order)
    }
    __syncthreads();

    v8f acc[8] = {};                           // [tile0: i,f,g,o][tile1: i,f,g,o]
    union AF { v16bf v; v8bf h[2]; };

    const __bf16* arowX0 = X   + (size_t)(m0a + lm) * HH;
    const __bf16* arowX1 = X   + (size_t)(m0b + lm) * HH;
    const __bf16* arowH0 = Hin + (size_t)(m0a + lm) * HH;
    const __bf16* arowH1 = Hin + (size_t)(m0b + lm) * HH;

    // ---------------- phase 1: X @ Wih^T (Whh DMA still in flight) --------
#pragma unroll
    for (int ks = 0; ks < 8; ++ks) {
        const int k0 = ks * 32;
        AF a0, a1;
        a0.h[0] = *(const v8bf*)(arowX0 + k0 + lh * 16);
        a0.h[1] = *(const v8bf*)(arowX0 + k0 + lh * 16 + 8);
        a1.h[0] = *(const v8bf*)(arowX1 + k0 + lh * 16);
        a1.h[1] = *(const v8bf*)(arowX1 + k0 + lh * 16 + 8);
        v16bf bfr[4];
#pragma unroll
        for (int g = 0; g < 4; ++g)
            bfr[g] = *(const v16bf*)(ldsW0 + (size_t)(g * 32 + nl) * HH + k0 + lh * 16);
#pragma unroll
        for (int g = 0; g < 4; ++g) {
            acc[g] = __builtin_amdgcn_wmma_f32_16x16x32_bf16(
                false, a0.v, false, bfr[g], (short)0, acc[g], false, false);
            acc[4 + g] = __builtin_amdgcn_wmma_f32_16x16x32_bf16(
                false, a1.v, false, bfr[g], (short)0, acc[4 + g], false, false);
        }
    }

    // ---------------- phase 2: Hin @ Whh^T --------------------------------
    if (wave_u == 0) __builtin_amdgcn_s_wait_tensorcnt(0);
    __syncthreads();
#pragma unroll
    for (int ks = 0; ks < 8; ++ks) {
        const int k0 = ks * 32;
        AF a0, a1;
        a0.h[0] = *(const v8bf*)(arowH0 + k0 + lh * 16);
        a0.h[1] = *(const v8bf*)(arowH0 + k0 + lh * 16 + 8);
        a1.h[0] = *(const v8bf*)(arowH1 + k0 + lh * 16);
        a1.h[1] = *(const v8bf*)(arowH1 + k0 + lh * 16 + 8);
        v16bf bfr[4];
#pragma unroll
        for (int g = 0; g < 4; ++g)
            bfr[g] = *(const v16bf*)(ldsW1 + (size_t)(g * 32 + nl) * HH + k0 + lh * 16);
#pragma unroll
        for (int g = 0; g < 4; ++g) {
            acc[g] = __builtin_amdgcn_wmma_f32_16x16x32_bf16(
                false, a0.v, false, bfr[g], (short)0, acc[g], false, false);
            acc[4 + g] = __builtin_amdgcn_wmma_f32_16x16x32_bf16(
                false, a1.v, false, bfr[g], (short)0, acc[4 + g], false, false);
        }
    }

    // --------- fused gate nonlinearity + state update (per-lane) ---------
    const float bi = bias[n];
    const float bf_ = bias[HH + n];
    const float bg = bias[2 * HH + n];
    const float bo = bias[3 * HH + n];
#pragma unroll
    for (int tile = 0; tile < 2; ++tile) {
        const int mbase = tile ? m0b : m0a;
        v8f* A = acc + tile * 4;
#pragma unroll
        for (int r = 0; r < 8; ++r) {
            const int m = mbase + r + lh * 8;      // C/D layout: lanes>=16 -> M+8
            const size_t off = (size_t)m * HH + n;
            const float iv = sigf(A[0][r] + bi);
            const float fv = sigf(A[1][r] + bf_);
            const float gv = tanhf(A[2][r] + bg);
            const float ov = sigf(A[3][r] + bo);
            const float cn = fv * C[off] + iv * gv;
            C[off] = cn;
            Hout[off] = (__bf16)(ov * tanhf(cn));
        }
    }
}

// ---------------------------------------------------------------------------
// Decoder step t: out[b][t][f] = prev + h2[b][:] . Wdec[f][:] + bdec[f]
// prev = x[b][9][f] (t==0) or out[b][t-1][f]  (carries the cumsum).
// ---------------------------------------------------------------------------
__global__ __launch_bounds__(256) void decode_step(
    const __bf16* __restrict__ h2, const float* __restrict__ Wdec,
    const float* __restrict__ bdec, const float* __restrict__ x,
    float* __restrict__ out, int t)
{
    int idx = blockIdx.x * 256 + threadIdx.x;
    if (idx >= BSZ * FF) return;
    int b = idx / FF, f = idx % FF;
    const __bf16* hr = h2 + (size_t)b * HH;
    const float*  wr = Wdec + (size_t)f * HH;
    float s = bdec[f];
#pragma unroll
    for (int kc = 0; kc < HH / 8; ++kc) {
        v8bf hv = *(const v8bf*)(hr + kc * 8);
#pragma unroll
        for (int j = 0; j < 8; ++j) s += (float)hv[j] * wr[kc * 8 + j];
    }
    float prev = (t == 0) ? x[((size_t)b * TT + (TT - 1)) * FF + f]
                          : out[((size_t)b * TT + (t - 1)) * FF + f];
    out[((size_t)b * TT + t) * FF + f] = prev + s;
}

// ---------------------------------------------------------------------------
extern "C" void kernel_launch(void* const* d_in, const int* in_sizes, int n_in,
                              void* d_out, int out_size, void* d_ws, size_t ws_size,
                              hipStream_t stream)
{
    const float* x    = (const float*)d_in[0];
    const float* Wenc = (const float*)d_in[1];
    const float* benc = (const float*)d_in[2];
    const float* Wih1 = (const float*)d_in[3];
    const float* Whh1 = (const float*)d_in[4];
    const float* bih1 = (const float*)d_in[5];
    const float* bhh1 = (const float*)d_in[6];
    const float* Wih2 = (const float*)d_in[7];
    const float* Whh2 = (const float*)d_in[8];
    const float* bih2 = (const float*)d_in[9];
    const float* bhh2 = (const float*)d_in[10];
    const float* Wdec = (const float*)d_in[11];
    const float* bdec = (const float*)d_in[12];
    float* out = (float*)d_out;

    // ---- workspace carve-out (256B aligned) ----
    char* ws = (char*)d_ws;
    size_t off = 0;
    auto take = [&](size_t bytes) -> char* {
        char* p = ws + off;
        off += (bytes + 255) & ~(size_t)255;
        return p;
    };
    __bf16* wih1b = (__bf16*)take((size_t)G4 * HH * 2);
    __bf16* whh1b = (__bf16*)take((size_t)G4 * HH * 2);
    __bf16* wih2b = (__bf16*)take((size_t)G4 * HH * 2);
    __bf16* whh2b = (__bf16*)take((size_t)G4 * HH * 2);
    float*  bias1 = (float*)take((size_t)G4 * 4);
    float*  bias2 = (float*)take((size_t)G4 * 4);
    __bf16* enc   = (__bf16*)take((size_t)TT * BSZ * HH * 2);
    __bf16* h1a   = (__bf16*)take((size_t)BSZ * HH * 2);
    __bf16* h1b   = (__bf16*)take((size_t)BSZ * HH * 2);
    __bf16* h2a   = (__bf16*)take((size_t)BSZ * HH * 2);
    __bf16* h2b   = (__bf16*)take((size_t)BSZ * HH * 2);
    float*  c1    = (float*)take((size_t)BSZ * HH * 4);
    float*  c2    = (float*)take((size_t)BSZ * HH * 4);
    (void)ws_size; (void)in_sizes; (void)n_in; (void)out_size;

    // ---- prep: weights -> bf16, biases folded, states zeroed ----
    pack_weights<<<(G4 * HH + 255) / 256, 256, 0, stream>>>(
        Wih1, Whh1, Wih2, Whh2, wih1b, whh1b, wih2b, whh2b);
    combine_bias<<<(G4 + 255) / 256, 256, 0, stream>>>(
        bih1, bhh1, bih2, bhh2, bias1, bias2);
    zero_state<<<(BSZ * HH) / 256, 256, 0, stream>>>(h1a, h2a, c1, c2);

    // ---- encoder: all T steps up front ----
    encoder_kernel<<<TT * BSZ, 256, 0, stream>>>(x, Wenc, benc, enc);

    // ---- recurrence ----
    dim3 cgrid(BSZ / 128, HH / 32);        // (128, 8)
    const size_t lds_bytes = 2 * 128 * HH * sizeof(__bf16);   // 128 KB double buffer
    __bf16 *h1c = h1a, *h1n = h1b, *h2c = h2a, *h2n = h2b;

    for (int t = 0; t < TT; ++t) {
        lstm_cell<<<cgrid, 256, lds_bytes, stream>>>(
            enc + (size_t)t * BSZ * HH, h1c, h1n, c1, wih1b, whh1b, bias1);
        { __bf16* tmp = h1c; h1c = h1n; h1n = tmp; }
        lstm_cell<<<cgrid, 256, lds_bytes, stream>>>(
            h1c, h2c, h2n, c2, wih2b, whh2b, bias2);
        { __bf16* tmp = h2c; h2c = h2n; h2n = tmp; }
    }

    const int dgrid = (BSZ * FF + 255) / 256;
    decode_step<<<dgrid, 256, 0, stream>>>(h2c, Wdec, bdec, x, out, 0);

    for (int i = 0; i < FUT; ++i) {
        lstm_cell<<<cgrid, 256, lds_bytes, stream>>>(
            h2c, h1c, h1n, c1, wih1b, whh1b, bias1);
        { __bf16* tmp = h1c; h1c = h1n; h1n = tmp; }
        lstm_cell<<<cgrid, 256, lds_bytes, stream>>>(
            h1c, h2c, h2n, c2, wih2b, whh2b, bias2);
        { __bf16* tmp = h2c; h2c = h2n; h2n = tmp; }
        decode_step<<<dgrid, 256, 0, stream>>>(h2c, Wdec, bdec, x, out, i + 1);
    }
}
